// GroupedQueryAttention_88957362635573
// MI455X (gfx1250) — matrix operand order repaired and checked by reading, not compiled, not run on hardware
//
#include <hip/hip_runtime.h>

// ---------------- problem constants ----------------
#define BATCH   2
#define SLEN    2048
#define HID     2048
#define NHEADS  16
#define NKVH    4
#define HEADD   128
#define SOFT_SCALE 0.08838834764831845f   // 1/sqrt(128)

typedef __bf16 bf16;
typedef __attribute__((ext_vector_type(16))) __bf16 bf16x16;
typedef __attribute__((ext_vector_type(8)))  __bf16 bf16x8;
typedef __attribute__((ext_vector_type(8)))  float  f32x8;
typedef __attribute__((ext_vector_type(4)))  unsigned int u32x4;
typedef __attribute__((ext_vector_type(8)))  int i32x8;
typedef __attribute__((ext_vector_type(4)))  int i32x4;

// ---------------- TDM (Tensor Data Mover) support ----------------
#if defined(__has_builtin)
#if __has_builtin(__builtin_amdgcn_tensor_load_to_lds)
#define HAVE_TDM 1
#endif
#endif
#ifndef HAVE_TDM
#define HAVE_TDM 0
#endif

#if __has_include(<hip/amd_detail/amd_gfx1250_TDM.h>)
#define TDM_ARITY6 1      // amdgpu-toolchain / therock headers -> 6-arg builtin
#else
#define TDM_ARITY6 0      // ROCm 7.2 -> 5-arg builtin
#endif

#if HAVE_TDM
// Issue a 2D bf16 tile DMA: global (row-major, src_ld elements) -> LDS
// (tile rows stored contiguously, row stride = cols elements).
// D# packing per CDNA5 ISA ch.8: group0 = {count, lds_addr, global_addr, type},
// group1 = {data_size, tensor_dim0/1, tile_dim0/1, tensor_dim0_stride}.
__device__ __forceinline__ void tdm_issue_2d(void* lds_dst, const void* gsrc,
                                             unsigned cols, unsigned rows,
                                             unsigned tensor_d0, unsigned tensor_d1,
                                             unsigned src_ld) {
  unsigned long long ga = (unsigned long long)gsrc;
  unsigned lds_off = (unsigned)(unsigned long long)lds_dst;   // LDS aperture low bits
  u32x4 g0;
  g0[0] = 1u;                                                  // count=1, user mode
  g0[1] = lds_off;                                             // lds_addr
  g0[2] = (unsigned)(ga & 0xffffffffu);                        // global_addr[31:0]
  g0[3] = (unsigned)((ga >> 32) & 0x01ffffffu) | (2u << 30);   // addr[56:32] | type=2
  i32x8 g1;
  g1[0] = (int)(1u << 16);                                     // wg_mask=0, data_size=1 (2B)
  g1[1] = (int)(tensor_d0 << 16);                              // tensor_dim0[15:0]
  g1[2] = (int)((tensor_d0 >> 16) | (tensor_d1 << 16));        // dim0[31:16] | dim1[15:0]
  g1[3] = (int)((tensor_d1 >> 16) | (cols << 16));             // dim1[31:16] | tile_dim0
  g1[4] = (int)rows;                                           // tile_dim1 | tile_dim2=0
  g1[5] = (int)src_ld;                                         // tensor_dim0_stride[31:0]
  g1[6] = 0;                                                   // stride[47:32] | dim1_stride lo
  g1[7] = 0;
  i32x4 z4 = {0, 0, 0, 0};
#if TDM_ARITY6
  i32x8 z8 = {0, 0, 0, 0, 0, 0, 0, 0};
  __builtin_amdgcn_tensor_load_to_lds(g0, g1, z4, z4, z8, 0);
#else
  __builtin_amdgcn_tensor_load_to_lds(g0, g1, z4, z4, 0);
#endif
}
#define TDM_WAIT(n) __builtin_amdgcn_s_wait_tensorcnt(n)
#else
#define TDM_WAIT(n)
#endif

// Stage a rows x cols bf16 tile into LDS. Call from ALL threads of the block.
// TDM path: single issue from wave 0 (EXEC-independent DMA). Fallback: co-op copy.
__device__ __forceinline__ void stage_tile(bf16* dst, const bf16* src,
                                           int rows, int cols, int src_ld,
                                           int tensor_d0, int tensor_d1) {
#if HAVE_TDM
  if (threadIdx.x == 0)
    tdm_issue_2d(dst, src, (unsigned)cols, (unsigned)rows,
                 (unsigned)tensor_d0, (unsigned)tensor_d1, (unsigned)src_ld);
#else
  const int chunks = rows * cols / 8;          // 16B chunks
  for (int i = threadIdx.x; i < chunks; i += blockDim.x) {
    int r = (i * 8) / cols, c = (i * 8) % cols;
    *(bf16x8*)(dst + r * cols + c) = *(const bf16x8*)(src + (size_t)r * src_ld + c);
  }
#endif
}

// ---------------- WMMA helpers ----------------
__device__ __forceinline__ f32x8 wmma_bf16(bf16x16 a, bf16x16 b, f32x8 c) {
  return __builtin_amdgcn_wmma_f32_16x16x32_bf16(false, a, false, b, (short)0, c,
                                                 false, false);
}

// A-matrix 16x32 bf16 fragment (lane<16: M=lane, K {0..7,16..23}; lane>=16: K {8..15,24..31})
__device__ __forceinline__ bf16x16 frag_a(const bf16* p, int ld, int r0, int k0, int lane) {
  const int m  = lane & 15;
  const int kb = (lane >> 4) << 3;
  const bf16* base = p + (size_t)(r0 + m) * ld + k0 + kb;
  bf16x16 f;
  ((bf16x8*)&f)[0] = *(const bf16x8*)(base);
  ((bf16x8*)&f)[1] = *(const bf16x8*)(base + 16);
  return f;
}

// B-matrix 32x16 fragment: B[k][n] = p[(n0+n)*ld + k0+k]
__device__ __forceinline__ bf16x16 frag_b(const bf16* p, int ld, int n0, int k0, int lane) {
  const int n  = lane & 15;
  const int kb = (lane >> 4) << 3;
  const bf16* base = p + (size_t)(n0 + n) * ld + k0 + kb;
  bf16x16 f;
  ((bf16x8*)&f)[0] = *(const bf16x8*)(base);
  ((bf16x8*)&f)[1] = *(const bf16x8*)(base + 16);
  return f;
}

__device__ __forceinline__ float hmax16(float v) {
  v = fmaxf(v, __shfl_xor(v, 1));
  v = fmaxf(v, __shfl_xor(v, 2));
  v = fmaxf(v, __shfl_xor(v, 4));
  v = fmaxf(v, __shfl_xor(v, 8));
  return v;
}
__device__ __forceinline__ float hsum16(float v) {
  v += __shfl_xor(v, 1);
  v += __shfl_xor(v, 2);
  v += __shfl_xor(v, 4);
  v += __shfl_xor(v, 8);
  return v;
}

// ---------------- kernel 0: f32 -> bf16 convert ----------------
__global__ void cvt_bf16_kernel(const float* __restrict__ in, bf16* __restrict__ out, int n) {
  int i = (blockIdx.x * blockDim.x + threadIdx.x) * 4;
  if (i + 3 < n) {
    float4 f = *(const float4*)(in + i);
    out[i + 0] = (bf16)f.x;
    out[i + 1] = (bf16)f.y;
    out[i + 2] = (bf16)f.z;
    out[i + 3] = (bf16)f.w;
  }
}

// ---------------- kernel 1: fused QKV projection + RoPE ----------------
// grid.x = (B*S)/64, grid.y = 24 (16 Q heads, 4 K, 4 V); 4 waves/block.
// Weight tile (128 out-rows x 64 K) staged into double-buffered LDS by TDM,
// shared by all 4 waves; each wave computes 16 token-rows x 128 cols (one head).
__global__ __launch_bounds__(128) void qkv_rope_kernel(
    const bf16* __restrict__ xb, const bf16* __restrict__ wqb,
    const bf16* __restrict__ wkb, const bf16* __restrict__ wvb,
    const float* __restrict__ cosT, const float* __restrict__ sinT,
    bf16* __restrict__ Qb, bf16* __restrict__ Kb, bf16* __restrict__ Vtb)
{
  const int lane = threadIdx.x & 31;
  const int wave = threadIdx.x >> 5;
  const int hi   = lane >> 4;
  const int nlo  = lane & 15;
  const int m0   = blockIdx.x * 64 + wave * 16;

  const int chunk = blockIdx.y;
  const bf16* W;
  int head, mode;                 // 0=Q(rope) 1=K(rope) 2=V(transposed)
  if (chunk < 16)      { W = wqb; head = chunk;      mode = 0; }
  else if (chunk < 20) { W = wkb; head = chunk - 16; mode = 1; }
  else                 { W = wvb; head = chunk - 20; mode = 2; }
  const size_t wbase = (size_t)head * HEADD * HID;

  __shared__ __align__(256) bf16 wtile[2][128 * 64];   // 2 x 16 KB

  const int nsteps = HID / 64;                          // 32
  stage_tile(wtile[0], W + wbase, 128, 64, HID, HID, NHEADS * HEADD);

  f32x8 acc[8] = {};
  for (int st = 0; st < nsteps; ++st) {
    __syncthreads();                                    // readers done with other buffer
    if (st + 1 < nsteps) {
      stage_tile(wtile[(st + 1) & 1], W + wbase + (st + 1) * 64,
                 128, 64, HID, HID, NHEADS * HEADD);
      TDM_WAIT(1);                                      // current tile complete
    } else {
      TDM_WAIT(0);
    }
    __syncthreads();                                    // tile visible to all waves
    const bf16* wt = wtile[st & 1];
#pragma unroll
    for (int kk = 0; kk < 64; kk += 32) {
      bf16x16 a = frag_a(xb, HID, m0, st * 64 + kk, lane);
#pragma unroll
      for (int j = 0; j < 8; ++j)
        acc[j] = wmma_bf16(a, frag_b(wt, 64, j * 16, kk, lane), acc[j]);
    }
  }

#pragma unroll
  for (int j = 0; j < 8; ++j) {
    const int d = j * 16 + nlo;
#pragma unroll
    for (int i = 0; i < 8; ++i) {
      const int t = m0 + i + 8 * hi;
      const int b = t >> 11;
      const int s = t & (SLEN - 1);
      const float v = acc[j][i];
      if (mode == 2) {
        Vtb[(((size_t)b * NKVH + head) * HEADD + d) * SLEN + s] = (bf16)v;
      } else {
        const float c  = cosT[s * HEADD + d];
        const float sn = sinT[s * HEADD + d];
        const float rot = (j < 4) ? -acc[j + 4][i] : acc[j - 4][i];
        const float o = v * c + rot * sn;
        if (mode == 0)
          Qb[(((size_t)b * NHEADS + head) * SLEN + s) * HEADD + d] = (bf16)o;
        else
          Kb[(((size_t)b * NKVH + head) * SLEN + s) * HEADD + d] = (bf16)o;
      }
    }
  }
}

// ---------------- kernel 2: causal flash attention ----------------
// grid.x = S/64, grid.y = B*NHEADS; 4 waves/block, each wave owns 16 queries.
// K and V 32-key tiles staged into double-buffered LDS via TDM (shared by all
// waves); online softmax; P transposed C->A layout through per-wave LDS.
__global__ __launch_bounds__(128) void flash_attn_kernel(
    const bf16* __restrict__ Qb, const bf16* __restrict__ Kb,
    const bf16* __restrict__ Vtb, bf16* __restrict__ ctxb)
{
  const int lane = threadIdx.x & 31;
  const int wave = threadIdx.x >> 5;
  const int hi   = lane >> 4;
  const int nlo  = lane & 15;

  const int q0  = blockIdx.x * 64 + wave * 16;
  const int bh  = blockIdx.y;
  const int b   = bh >> 4;
  const int h   = bh & 15;
  const int kvh = h >> 2;

  const bf16* Qh  = Qb  + ((size_t)(b * NHEADS + h)) * SLEN * HEADD;
  const bf16* Kh  = Kb  + ((size_t)(b * NKVH + kvh)) * SLEN * HEADD;
  const bf16* Vth = Vtb + ((size_t)(b * NKVH + kvh)) * HEADD * SLEN;

  __shared__ __align__(256) bf16 ldsK[2][32 * 128];    // keys x d     (2 x 8 KB)
  __shared__ __align__(256) bf16 ldsV[2][128 * 32];    // d x keys     (2 x 8 KB)
  __shared__ __align__(32)  bf16 pb[4][16][32];        // per-wave P transpose

  bf16x16 qf[4];
#pragma unroll
  for (int t = 0; t < 4; ++t) qf[t] = frag_a(Qh, HEADD, q0, t * 32, lane);

  f32x8 ctx[8] = {};
  float m[8], l[8];
#pragma unroll
  for (int i = 0; i < 8; ++i) { m[i] = -3.0e38f; l[i] = 0.0f; }

  const int nkb = 2 * blockIdx.x + 2;   // block-uniform trip count (causal edge)

  stage_tile(ldsK[0], Kh, 32, 128, HEADD, HEADD, SLEN);       // keys 0..31
  stage_tile(ldsV[0], Vth, 128, 32, SLEN, SLEN, HEADD);

  for (int kb = 0; kb < nkb; ++kb) {
    const int kk = kb * 32;
    __syncthreads();                    // all waves done with the other buffers
    if (kb + 1 < nkb) {
      const int nk = kk + 32;
      stage_tile(ldsK[(kb + 1) & 1], Kh + (size_t)nk * HEADD, 32, 128, HEADD,
                 HEADD, SLEN);
      stage_tile(ldsV[(kb + 1) & 1], Vth + nk, 128, 32, SLEN, SLEN, HEADD);
      TDM_WAIT(2);                      // current K/V pair complete, next in flight
    } else {
      TDM_WAIT(0);
    }
    __syncthreads();                    // staged tiles visible
    const bf16* kt = ldsK[kb & 1];
    const bf16* vt = ldsV[kb & 1];

    // scores S[16q x 32k] = Q * K^T
    f32x8 s0 = {}, s1 = {};
#pragma unroll
    for (int t = 0; t < 4; ++t) {
      s0 = wmma_bf16(qf[t], frag_b(kt, 128, 0,  t * 32, lane), s0);
      s1 = wmma_bf16(qf[t], frag_b(kt, 128, 16, t * 32, lane), s1);
    }

    float p0[8], p1[8], alpha[8];
#pragma unroll
    for (int i = 0; i < 8; ++i) {
      const int q = q0 + i + 8 * hi;
      float a0 = s0[i] * SOFT_SCALE + ((kk + nlo)      > q ? -1e9f : 0.0f);
      float a1 = s1[i] * SOFT_SCALE + ((kk + 16 + nlo) > q ? -1e9f : 0.0f);
      const float mx   = hmax16(fmaxf(a0, a1));
      const float mnew = fmaxf(m[i], mx);
      const float al   = __expf(m[i] - mnew);
      const float e0 = __expf(a0 - mnew);
      const float e1 = __expf(a1 - mnew);
      l[i] = l[i] * al + hsum16(e0 + e1);
      m[i] = mnew;
      alpha[i] = al; p0[i] = e0; p1[i] = e1;
    }
#pragma unroll
    for (int j = 0; j < 8; ++j)
#pragma unroll
      for (int i = 0; i < 8; ++i) ctx[j][i] *= alpha[i];

    // P: C-layout -> A-layout through per-wave LDS
#pragma unroll
    for (int i = 0; i < 8; ++i) {
      const int r = i + 8 * hi;
      pb[wave][r][nlo]      = (bf16)p0[i];
      pb[wave][r][16 + nlo] = (bf16)p1[i];
    }
    __syncthreads();
    const bf16x16 pa = frag_a(&pb[wave][0][0], 32, 0, 0, lane);

    // ctx[16q x 128d] += P[16x32] * V[32x128]  (V tile stored [d][keys])
#pragma unroll
    for (int j = 0; j < 8; ++j)
      ctx[j] = wmma_bf16(pa, frag_b(vt, 32, j * 16, 0, lane), ctx[j]);
  }

  float inv[8];
#pragma unroll
  for (int i = 0; i < 8; ++i) inv[i] = 1.0f / l[i];
#pragma unroll
  for (int j = 0; j < 8; ++j)
#pragma unroll
    for (int i = 0; i < 8; ++i) {
      const int q = q0 + i + 8 * hi;
      ctxb[((size_t)(b * SLEN + q)) * (NHEADS * HEADD) + h * HEADD + j * 16 + nlo] =
          (bf16)(ctx[j][i] * inv[i]);
    }
}

// ---------------- kernel 3: output projection (f32 out) ----------------
// grid.x = (B*S)/64, grid.y = HID/64; Wo tile (64 x 64K) TDM-staged, shared.
__global__ __launch_bounds__(128) void out_proj_kernel(
    const bf16* __restrict__ ctxb, const bf16* __restrict__ wob,
    float* __restrict__ out)
{
  const int lane = threadIdx.x & 31;
  const int wave = threadIdx.x >> 5;
  const int hi   = lane >> 4;
  const int nlo  = lane & 15;
  const int m0 = blockIdx.x * 64 + wave * 16;
  const int n0 = blockIdx.y * 64;

  __shared__ __align__(256) bf16 wtile[2][64 * 64];    // 2 x 8 KB

  const int nsteps = HID / 64;
  stage_tile(wtile[0], wob + (size_t)n0 * HID, 64, 64, HID, HID, HID);

  f32x8 acc[4] = {};
  for (int st = 0; st < nsteps; ++st) {
    __syncthreads();
    if (st + 1 < nsteps) {
      stage_tile(wtile[(st + 1) & 1], wob + (size_t)n0 * HID + (st + 1) * 64,
                 64, 64, HID, HID, HID);
      TDM_WAIT(1);
    } else {
      TDM_WAIT(0);
    }
    __syncthreads();
    const bf16* wt = wtile[st & 1];
#pragma unroll
    for (int kk = 0; kk < 64; kk += 32) {
      bf16x16 a = frag_a(ctxb, HID, m0, st * 64 + kk, lane);
#pragma unroll
      for (int j = 0; j < 4; ++j)
        acc[j] = wmma_bf16(a, frag_b(wt, 64, j * 16, kk, lane), acc[j]);
    }
  }
#pragma unroll
  for (int j = 0; j < 4; ++j)
#pragma unroll
    for (int i = 0; i < 8; ++i)
      out[(size_t)(m0 + i + 8 * hi) * HID + n0 + j * 16 + nlo] = acc[j][i];
}

// ---------------- host launcher ----------------
extern "C" void kernel_launch(void* const* d_in, const int* in_sizes, int n_in,
                              void* d_out, int out_size, void* d_ws, size_t ws_size,
                              hipStream_t stream) {
  (void)in_sizes; (void)n_in; (void)out_size; (void)ws_size;
  const float* x    = (const float*)d_in[0];
  const float* Wq   = (const float*)d_in[1];
  const float* Wk   = (const float*)d_in[2];
  const float* Wv   = (const float*)d_in[3];
  const float* Wo   = (const float*)d_in[4];
  const float* cosT = (const float*)d_in[5];
  const float* sinT = (const float*)d_in[6];
  float* out = (float*)d_out;   // mask input d_in[7] is redundant (causal analytic)

  const size_t NX  = (size_t)BATCH * SLEN * HID;
  const size_t NWQ = (size_t)NHEADS * HEADD * HID;
  const size_t NWK = (size_t)NKVH * HEADD * HID;
  const size_t NQ  = (size_t)BATCH * NHEADS * SLEN * HEADD;
  const size_t NKV = (size_t)BATCH * NKVH * SLEN * HEADD;

  char* ws = (char*)d_ws;
  size_t off = 0;
  bf16* xb  = (bf16*)(ws + off); off += NX  * 2;
  bf16* wqb = (bf16*)(ws + off); off += NWQ * 2;
  bf16* wkb = (bf16*)(ws + off); off += NWK * 2;
  bf16* wvb = (bf16*)(ws + off); off += NWK * 2;
  bf16* wob = (bf16*)(ws + off); off += NWQ * 2;
  bf16* Qb  = (bf16*)(ws + off); off += NQ  * 2;
  bf16* Kb  = (bf16*)(ws + off); off += NKV * 2;
  bf16* Vtb = (bf16*)(ws + off); off += NKV * 2;
  bf16* ctxb= (bf16*)(ws + off); off += NX  * 2;   // ~76 MB total

  auto cvt = [&](const float* src, bf16* dst, size_t n) {
    int blocks = (int)((n / 4 + 255) / 256);
    cvt_bf16_kernel<<<blocks, 256, 0, stream>>>(src, dst, (int)n);
  };
  cvt(x,  xb,  NX);
  cvt(Wq, wqb, NWQ);
  cvt(Wk, wkb, NWK);
  cvt(Wv, wvb, NWK);
  cvt(Wo, wob, NWQ);

  qkv_rope_kernel<<<dim3((BATCH * SLEN) / 64, 24), 128, 0, stream>>>(
      xb, wqb, wkb, wvb, cosT, sinT, Qb, Kb, Vtb);

  flash_attn_kernel<<<dim3(SLEN / 64, BATCH * NHEADS), 128, 0, stream>>>(
      Qb, Kb, Vtb, ctxb);

  out_proj_kernel<<<dim3((BATCH * SLEN) / 64, HID / 64), 128, 0, stream>>>(
      ctxb, wob, out);
}
